// LSNN_layer_47614007444119
// MI455X (gfx1250) — matrix operand order repaired and checked by reading, not compiled, not run on hardware
//
#include <hip/hip_runtime.h>
#include <hip/hip_bf16.h>
#include <stdint.h>

// Problem dims (fixed by the reference)
#define B_DIM 4096
#define I_DIM 2048
#define O_DIM 2048
#define LDG   2048   // leading dim of every operand matrix (I==O==2048)

typedef __attribute__((ext_vector_type(16))) __bf16 v16bf;
typedef __attribute__((ext_vector_type(8)))  float  v8f;

union FragU {
  uint4 q[2];            // 32 bytes
  v16bf v;
  unsigned short s[16];
};

// round-to-nearest-even f32 -> bf16 (bit pattern)
__device__ __forceinline__ unsigned short f2bf(float f) {
  unsigned int u = __float_as_uint(f);
  u += 0x7FFFu + ((u >> 16) & 1u);
  return (unsigned short)(u >> 16);
}

// Vectorized f32 -> bf16: 4 floats / thread (float4 in, uint2 out)
__global__ void f32_to_bf16_kernel(const float4* __restrict__ src,
                                   uint2* __restrict__ dst, int n4) {
  int i = blockIdx.x * blockDim.x + threadIdx.x;
  if (i < n4) {
    float4 v = src[i];
    uint2 o;
    o.x = (unsigned)f2bf(v.x) | ((unsigned)f2bf(v.y) << 16);
    o.y = (unsigned)f2bf(v.z) | ((unsigned)f2bf(v.w) << 16);
    dst[i] = o;
  }
}

// ---------------- async global -> LDS staging (CDNA5 path) ----------------
#define LDT 40   // padded LDS row stride in bf16 elements (32 data + 8 pad)

// One 16B chunk, ASYNCcnt-tracked, no VGPR round trip.
__device__ __forceinline__ void async_cp16(unsigned short* ldsdst,
                                           const unsigned short* src) {
  unsigned int l = (unsigned int)(uintptr_t)ldsdst;  // low 32 bits = LDS offset
  asm volatile("global_load_async_to_lds_b128 %0, %1, off"
               :
               : "v"(l), "v"(src)
               : "memory");
}

__device__ __forceinline__ void wait_async0() {
  asm volatile("s_wait_asynccnt 0" ::: "memory");
}

// 128 x 32 bf16 tile: 512 chunks, 2 per thread (256 threads)
__device__ __forceinline__ void stage_tileA(unsigned short* T,
                                            const unsigned short* __restrict__ g,
                                            int row0, int k0, int tid) {
  int r  = tid >> 2;
  int k8 = (tid & 3) << 3;
  async_cp16(T + r * LDT + k8,        g + (size_t)(row0 + r)      * LDG + k0 + k8);
  async_cp16(T + (r + 64) * LDT + k8, g + (size_t)(row0 + r + 64) * LDG + k0 + k8);
}

// 64 x 32 bf16 tile: 256 chunks, 1 per thread
__device__ __forceinline__ void stage_tileB(unsigned short* T,
                                            const unsigned short* __restrict__ g,
                                            int row0, int k0, int tid) {
  int r  = tid >> 2;
  int k8 = (tid & 3) << 3;
  async_cp16(T + r * LDT + k8, g + (size_t)(row0 + r) * LDG + k0 + k8);
}

// ---------------- WMMA fragment loads from LDS ----------------
// A-fragment (16x32 bf16, M x K). lane<16: K = 0..7 & 16..23; lane>=16: +8.
__device__ __forceinline__ v16bf load_a_frag(const unsigned short* lds,
                                             int rowbase, int lane) {
  int kb = (lane >> 4) << 3;
  const unsigned short* p = lds + rowbase * LDT + kb;
  FragU f;
  f.q[0] = *(const uint4*)(p);
  f.q[1] = *(const uint4*)(p + 16);
  return f.v;
}

// B-fragment (32x16 bf16, K x N), W[n,k] row-major. lane=N; K = 0..7 & 16..23; lane>=16: +8.
__device__ __forceinline__ v16bf load_b_frag(const unsigned short* lds,
                                             int nbase, int lane) {
  int ks = (lane >> 4) << 3;
  const unsigned short* p = lds + nbase * LDT + ks;
  FragU f;
  f.q[0] = *(const uint4*)(p);
  f.q[1] = *(const uint4*)(p + 16);
  return f.v;
}

__device__ __forceinline__ v8f wmma_bf16(v16bf a, v16bf b, v8f c) {
  return __builtin_amdgcn_wmma_f32_16x16x32_bf16(
      /*neg_a=*/false, a, /*neg_b=*/false, b,
      /*c_mod=*/(short)0, c, /*reuse_a=*/false, /*reuse_b=*/false);
}

// ---------------- GEMM1: L1 = x @ W_syn^T + b_syn ----------------
__global__ __launch_bounds__(256)
void gemm1_kernel(const unsigned short* __restrict__ Abf,  // x bf16 [B,I]
                  const unsigned short* __restrict__ Wbf,  // W_syn bf16 [O,I]
                  const float* __restrict__ bias,          // b_syn [O]
                  const float* __restrict__ u_t,           // [B,O]
                  const float* __restrict__ b_t,           // [B,O]
                  float* __restrict__ L1out,               // [B,O] fp32
                  unsigned short* __restrict__ A2bf,       // bf16(L1+u)
                  unsigned short* __restrict__ A3bf) {     // bf16(L1+b)
  __shared__ alignas(16) unsigned short AT[2][128 * LDT];
  __shared__ alignas(16) unsigned short BT[2][64 * LDT];

  int tid  = threadIdx.x;
  int lane = tid & 31;
  int wave = tid >> 5;                 // 8 waves (wave32)
  int waveM = (wave & 3) * 32;         // 4 row groups
  int waveN = (wave >> 2) * 32;        // 2 col groups
  int m0 = blockIdx.y * 128;
  int n0 = blockIdx.x * 64;

  v8f acc[2][2] = {};

  // prologue: stage first K-tile into buffer 0
  stage_tileA(AT[0], Abf, m0, 0, tid);
  stage_tileB(BT[0], Wbf, n0, 0, tid);

  int buf = 0;
  for (int k0 = 0; k0 < LDG; k0 += 32, buf ^= 1) {
    wait_async0();        // my loads for buf are done
    __syncthreads();      // everyone's loads done + other buffer free
    if (k0 + 32 < LDG) {  // stage next tile into other buffer (overlaps compute)
      stage_tileA(AT[buf ^ 1], Abf, m0, k0 + 32, tid);
      stage_tileB(BT[buf ^ 1], Wbf, n0, k0 + 32, tid);
    }
    const unsigned short* ATc = AT[buf];
    const unsigned short* BTc = BT[buf];

    v16bf aF[2], bF[2];
#pragma unroll
    for (int tm = 0; tm < 2; ++tm)
      aF[tm] = load_a_frag(ATc, waveM + tm * 16 + (lane & 15), lane);
#pragma unroll
    for (int tn = 0; tn < 2; ++tn)
      bF[tn] = load_b_frag(BTc, waveN + tn * 16 + (lane & 15), lane);
#pragma unroll
    for (int tm = 0; tm < 2; ++tm)
#pragma unroll
      for (int tn = 0; tn < 2; ++tn)
        acc[tm][tn] = wmma_bf16(aF[tm], bF[tn], acc[tm][tn]);
  }

  // Epilogue: element (vgpr r, lane L): M = r + (L>>4)*8, N = L&15
  int rofs = (lane >> 4) << 3;
  int cofs = lane & 15;
#pragma unroll
  for (int tm = 0; tm < 2; ++tm)
#pragma unroll
    for (int tn = 0; tn < 2; ++tn)
#pragma unroll
      for (int r = 0; r < 8; ++r) {
        int row = m0 + waveM + tm * 16 + rofs + r;
        int col = n0 + waveN + tn * 16 + cofs;
        size_t idx = (size_t)row * O_DIM + col;
        float l1 = acc[tm][tn][r] + bias[col];
        L1out[idx] = l1;
        A2bf[idx]  = f2bf(l1 + u_t[idx]);
        A3bf[idx]  = f2bf(l1 + b_t[idx]);
      }
}

// ---- GEMM2 (fused dual GEMM + full LSNN epilogue -> spikes) ----
__global__ __launch_bounds__(256)
void gemm2_fused_kernel(const unsigned short* __restrict__ A2bf, // [B,O]
                        const unsigned short* __restrict__ A3bf, // [B,O]
                        const unsigned short* __restrict__ Wm,   // W_Tm bf16 [O,O]
                        const unsigned short* __restrict__ Wa,   // W_Tadp bf16 [O,O]
                        const float* __restrict__ b_Tm,
                        const float* __restrict__ b_Tadp,
                        const float* __restrict__ L1,
                        const float* __restrict__ u_t,
                        const float* __restrict__ b_t,
                        const float* __restrict__ spk,
                        float* __restrict__ out) {
  __shared__ alignas(16) unsigned short AT2[2][128 * LDT];
  __shared__ alignas(16) unsigned short AT3[2][128 * LDT];
  __shared__ alignas(16) unsigned short BTm[2][64 * LDT];
  __shared__ alignas(16) unsigned short BTa[2][64 * LDT];

  int tid  = threadIdx.x;
  int lane = tid & 31;
  int wave = tid >> 5;
  int waveM = (wave & 3) * 32;
  int waveN = (wave >> 2) * 32;
  int m0 = blockIdx.y * 128;
  int n0 = blockIdx.x * 64;

  v8f acc2[2][2] = {};
  v8f acc3[2][2] = {};

  stage_tileA(AT2[0], A2bf, m0, 0, tid);
  stage_tileA(AT3[0], A3bf, m0, 0, tid);
  stage_tileB(BTm[0], Wm,   n0, 0, tid);
  stage_tileB(BTa[0], Wa,   n0, 0, tid);

  int buf = 0;
  for (int k0 = 0; k0 < LDG; k0 += 32, buf ^= 1) {
    wait_async0();
    __syncthreads();
    if (k0 + 32 < LDG) {
      stage_tileA(AT2[buf ^ 1], A2bf, m0, k0 + 32, tid);
      stage_tileA(AT3[buf ^ 1], A3bf, m0, k0 + 32, tid);
      stage_tileB(BTm[buf ^ 1], Wm,   n0, k0 + 32, tid);
      stage_tileB(BTa[buf ^ 1], Wa,   n0, k0 + 32, tid);
    }
    const unsigned short* A2c = AT2[buf];
    const unsigned short* A3c = AT3[buf];
    const unsigned short* Bmc = BTm[buf];
    const unsigned short* Bac = BTa[buf];

    v16bf a2F[2], a3F[2], bmF[2], baF[2];
#pragma unroll
    for (int tm = 0; tm < 2; ++tm) {
      a2F[tm] = load_a_frag(A2c, waveM + tm * 16 + (lane & 15), lane);
      a3F[tm] = load_a_frag(A3c, waveM + tm * 16 + (lane & 15), lane);
    }
#pragma unroll
    for (int tn = 0; tn < 2; ++tn) {
      bmF[tn] = load_b_frag(Bmc, waveN + tn * 16 + (lane & 15), lane);
      baF[tn] = load_b_frag(Bac, waveN + tn * 16 + (lane & 15), lane);
    }
#pragma unroll
    for (int tm = 0; tm < 2; ++tm)
#pragma unroll
      for (int tn = 0; tn < 2; ++tn) {
        acc2[tm][tn] = wmma_bf16(a2F[tm], bmF[tn], acc2[tm][tn]);
        acc3[tm][tn] = wmma_bf16(a3F[tm], baF[tn], acc3[tm][tn]);
      }
  }

  // LSNN epilogue — alpha/rho stay in registers, write spikes directly.
  int rofs = (lane >> 4) << 3;
  int cofs = lane & 15;
#pragma unroll
  for (int tm = 0; tm < 2; ++tm)
#pragma unroll
    for (int tn = 0; tn < 2; ++tn)
#pragma unroll
      for (int r = 0; r < 8; ++r) {
        int row = m0 + waveM + tm * 16 + rofs + r;
        int col = n0 + waveN + tn * 16 + cofs;
        size_t idx = (size_t)row * O_DIM + col;
        float p2 = acc2[tm][tn][r] + b_Tm[col];
        float p3 = acc3[tm][tn][r] + b_Tadp[col];
        float alpha = 1.0f / (1.0f + __expf(-p2));
        float rho   = 1.0f / (1.0f + __expf(-p3));
        float l1 = L1[idx];
        float u  = u_t[idx];
        float bt = b_t[idx];
        float s  = spk[idx];
        float bnew = rho * bt + (1.0f - rho) * s;
        float thr  = 0.01f + 1.8f * bnew;
        float unew = u + (l1 - u) / alpha;
        out[idx] = (unew - thr > 0.0f) ? 1.0f : 0.0f;
      }
}

extern "C" void kernel_launch(void* const* d_in, const int* in_sizes, int n_in,
                              void* d_out, int out_size, void* d_ws, size_t ws_size,
                              hipStream_t stream) {
  (void)in_sizes; (void)n_in; (void)out_size; (void)ws_size;

  const float* x_t    = (const float*)d_in[0];
  const float* u_t    = (const float*)d_in[1];
  const float* b_t    = (const float*)d_in[2];
  const float* spk    = (const float*)d_in[3];
  const float* W_syn  = (const float*)d_in[4];
  const float* b_syn  = (const float*)d_in[5];
  const float* W_Tm   = (const float*)d_in[6];
  const float* b_Tm   = (const float*)d_in[7];
  const float* W_Tadp = (const float*)d_in[8];
  const float* b_Tadp = (const float*)d_in[9];
  float* out = (float*)d_out;

  // Workspace layout (all 16B-aligned multiples)
  char* ws = (char*)d_ws;
  unsigned short* x_bf    = (unsigned short*)ws;  ws += (size_t)B_DIM * I_DIM * 2;
  unsigned short* wsyn_bf = (unsigned short*)ws;  ws += (size_t)O_DIM * I_DIM * 2;
  unsigned short* wtm_bf  = (unsigned short*)ws;  ws += (size_t)O_DIM * O_DIM * 2;
  unsigned short* wta_bf  = (unsigned short*)ws;  ws += (size_t)O_DIM * O_DIM * 2;
  unsigned short* A2bf    = (unsigned short*)ws;  ws += (size_t)B_DIM * O_DIM * 2;
  unsigned short* A3bf    = (unsigned short*)ws;  ws += (size_t)B_DIM * O_DIM * 2;
  float* L1ws             = (float*)ws;           // B*O fp32

  // Pass 0: f32 -> bf16 conversions (vectorized x4)
  {
    int n4 = (B_DIM * I_DIM) / 4;
    f32_to_bf16_kernel<<<(n4 + 255) / 256, 256, 0, stream>>>(
        (const float4*)x_t, (uint2*)x_bf, n4);
  }
  {
    int n4 = (O_DIM * I_DIM) / 4;
    f32_to_bf16_kernel<<<(n4 + 255) / 256, 256, 0, stream>>>(
        (const float4*)W_syn, (uint2*)wsyn_bf, n4);
  }
  {
    int n4 = (O_DIM * O_DIM) / 4;
    f32_to_bf16_kernel<<<(n4 + 255) / 256, 256, 0, stream>>>(
        (const float4*)W_Tm, (uint2*)wtm_bf, n4);
    f32_to_bf16_kernel<<<(n4 + 255) / 256, 256, 0, stream>>>(
        (const float4*)W_Tadp, (uint2*)wta_bf, n4);
  }

  dim3 grid(O_DIM / 64, B_DIM / 128);  // 32 x 32 macro-tiles

  // Pass 1: L1 + pre-converted phase-2 activations
  gemm1_kernel<<<grid, 256, 0, stream>>>(x_bf, wsyn_bf, b_syn, u_t, b_t,
                                         L1ws, A2bf, A3bf);

  // Pass 2: dual GEMM + fused spike epilogue
  gemm2_fused_kernel<<<grid, 256, 0, stream>>>(A2bf, A3bf, wtm_bf, wta_bf,
                                               b_Tm, b_Tadp, L1ws,
                                               u_t, b_t, spk, out);
}
